// Branch1_block_81630148428039
// MI455X (gfx1250) — matrix operand order repaired and checked by reading, not compiled, not run on hardware
//
#include <hip/hip_runtime.h>
#include <hip/hip_bf16.h>

typedef __attribute__((ext_vector_type(16))) __bf16 v16bf;
typedef __attribute__((ext_vector_type(8)))  __bf16 v8bf;
typedef __attribute__((ext_vector_type(4)))  __bf16 v4bf;
typedef __attribute__((ext_vector_type(8)))  float  v8f;

#define NB   2048
#define BB   32
#define TT_  12
#define FIN  16
#define FOUT 32

// ---------------------------------------------------------------------------
// SE attention: per-(b,t) mean over (f,n)  -> att_raw[384]
// ---------------------------------------------------------------------------
__global__ __launch_bounds__(256) void k_mean(const float* __restrict__ x,
                                              float* __restrict__ att_raw) {
    const int bt = blockIdx.x;
    const float4* p = (const float4*)(x + (size_t)bt * (FIN * NB));
    float s = 0.f;
    for (int i = threadIdx.x; i < (FIN * NB) / 4; i += 256) {
        float4 f = p[i];
        s += f.x + f.y + f.z + f.w;
    }
    __shared__ float red[256];
    red[threadIdx.x] = s;
    __syncthreads();
    for (int st = 128; st > 0; st >>= 1) {
        if (threadIdx.x < st) red[threadIdx.x] += red[threadIdx.x + st];
        __syncthreads();
    }
    if (threadIdx.x == 0) att_raw[bt] = red[0] * (1.f / (FIN * NB));
}

// SE MLP: 12 -> 3 (relu) -> 12 (sigmoid). One thread per batch b.
__global__ __launch_bounds__(32) void k_semlp(const float* __restrict__ att_raw,
                                              const float* __restrict__ w1,
                                              const float* __restrict__ b1,
                                              const float* __restrict__ w2,
                                              const float* __restrict__ b2,
                                              float* __restrict__ att) {
    const int b = threadIdx.x;
    if (b >= BB) return;
    float a[TT_];
#pragma unroll
    for (int t = 0; t < TT_; ++t) a[t] = att_raw[b * TT_ + t];
    float h[3];
#pragma unroll
    for (int j = 0; j < 3; ++j) {
        float s = b1[j];
#pragma unroll
        for (int t = 0; t < TT_; ++t) s += w1[j * TT_ + t] * a[t];
        h[j] = fmaxf(s, 0.f);
    }
#pragma unroll
    for (int t = 0; t < TT_; ++t) {
        float s = b2[t];
#pragma unroll
        for (int j = 0; j < 3; ++j) s += w2[t * 3 + j] * h[j];
        att[b * TT_ + t] = 1.f / (1.f + __expf(-s));
    }
}

// ---------------------------------------------------------------------------
// f32 -> bf16 converters
// ---------------------------------------------------------------------------
__global__ __launch_bounds__(256) void k_f32_to_bf16(const float* __restrict__ src,
                                                     __bf16* __restrict__ dst, int n4) {
    int i = blockIdx.x * 256 + threadIdx.x;
    if (i >= n4) return;
    float4 f = ((const float4*)src)[i];
    v4bf o;
    o[0] = (__bf16)f.x; o[1] = (__bf16)f.y; o[2] = (__bf16)f.z; o[3] = (__bf16)f.w;
    ((v4bf*)dst)[i] = o;
}

// h1 = bf16(x * att[b,t])
__global__ __launch_bounds__(256) void k_scale_to_bf16(const float* __restrict__ x,
                                                       const float* __restrict__ att,
                                                       __bf16* __restrict__ dst, int n4) {
    int i = blockIdx.x * 256 + threadIdx.x;
    if (i >= n4) return;
    const int bt = i / ((FIN * NB) / 4);
    const float a = att[bt];
    float4 f = ((const float4*)x)[i];
    v4bf o;
    o[0] = (__bf16)(f.x * a); o[1] = (__bf16)(f.y * a);
    o[2] = (__bf16)(f.z * a); o[3] = (__bf16)(f.w * a);
    ((v4bf*)dst)[i] = o;
}

// ---------------------------------------------------------------------------
// Fused Chebyshev graph conv:
//   out[bt,o,n] = relu( sum_{k,f} theta[k,f,o] * sum_m H[bt,f,m] * cheb[k,m,n] )
// cheb is SYMMETRIC, so B-frag lane loads are contiguous rows cheb[k][n][m..].
// Block: 8 waves x 16 columns = 128 columns of n.  Grid: (B*T) * (N/128).
// ---------------------------------------------------------------------------
template <int F>
__global__ __launch_bounds__(256) void k_gconv(const __bf16* __restrict__ H,
                                               const __bf16* __restrict__ Cb,
                                               const float* __restrict__ theta,
                                               __bf16* __restrict__ Out) {
    constexpr int FA = F / 16;          // A row-tiles
    constexpr int KF = 3 * F;           // contraction length of feature mix
    __shared__ float thLds[KF * 32];
    __shared__ float zscr[8][KF * 16];  // per-wave Z scratch (reused as out-bounce)

    for (int i = threadIdx.x; i < KF * 32; i += 256) thLds[i] = theta[i];
    __syncthreads();

    const int tile = blockIdx.x & 15;
    const int bt   = blockIdx.x >> 4;
    const int lane = threadIdx.x & 31;
    const int wv   = threadIdx.x >> 5;
    const int r    = lane & 15;         // A row / B column / D column within tile
    const int lh   = lane >> 4;         // lane half
    const int n0   = tile * 128 + wv * 16;
    const int ncol = n0 + r;

    // base pointers
    const __bf16* hrow[FA];
#pragma unroll
    for (int fa = 0; fa < FA; ++fa)
        hrow[fa] = H + ((size_t)bt * F + fa * 16 + r) * NB;
    const __bf16* bbase[3];
#pragma unroll
    for (int k = 0; k < 3; ++k)
        bbase[k] = Cb + (size_t)k * NB * NB + (size_t)ncol * NB;

    v8f acc[3][FA];
#pragma unroll
    for (int k = 0; k < 3; ++k)
#pragma unroll
        for (int fa = 0; fa < FA; ++fa)
#pragma unroll
            for (int i = 0; i < 8; ++i) acc[k][fa][i] = 0.f;

    for (int mc = 0; mc < NB / 32; ++mc) {
        const int m0 = mc * 32;
        __builtin_prefetch(bbase[0] + m0 + 512, 0, 0);  // global_prefetch_b8 on cheb stream

        // A fragments: ISA 16-bit A layout: K = {8*lh+0..7} and {16+8*lh+0..7}
        v16bf Af[FA];
#pragma unroll
        for (int fa = 0; fa < FA; ++fa) {
            const v8bf lo = *(const v8bf*)(hrow[fa] + m0 + lh * 8);
            const v8bf hi = *(const v8bf*)(hrow[fa] + m0 + 16 + lh * 8);
#pragma unroll
            for (int i = 0; i < 8; ++i) { Af[fa][i] = lo[i]; Af[fa][i + 8] = hi[i]; }
        }
#pragma unroll
        for (int k = 0; k < 3; ++k) {
            // B fragment: K = {8*lh+0..7} and {16+8*lh+0..7}, col = r; symmetric cheb -> contiguous
            const v8bf lo = *(const v8bf*)(bbase[k] + m0 + lh * 8);
            const v8bf hi = *(const v8bf*)(bbase[k] + m0 + 16 + lh * 8);
            v16bf Bf;
#pragma unroll
            for (int i = 0; i < 8; ++i) { Bf[i] = lo[i]; Bf[i + 8] = hi[i]; }
#pragma unroll
            for (int fa = 0; fa < FA; ++fa)
                acc[k][fa] = __builtin_amdgcn_wmma_f32_16x16x32_bf16(
                    false, Af[fa], false, Bf, (short)0, acc[k][fa], false, false);
        }
    }

    // Dump Z to LDS: D layout -> lane holds col r, rows v + 8*lh
#pragma unroll
    for (int k = 0; k < 3; ++k)
#pragma unroll
        for (int fa = 0; fa < FA; ++fa)
#pragma unroll
            for (int v = 0; v < 8; ++v)
                zscr[wv][(k * F + fa * 16 + v + 8 * lh) * 16 + r] = acc[k][fa][v];
    __syncthreads();

    // Feature mix in f32: lane covers column r, o-range [16*lh, 16*lh+16)
    float o16[16];
#pragma unroll
    for (int i = 0; i < 16; ++i) o16[i] = 0.f;
    for (int kf = 0; kf < KF; ++kf) {
        const float z = zscr[wv][kf * 16 + r];
        const float4* th4 = (const float4*)&thLds[kf * 32 + lh * 16];
#pragma unroll
        for (int q = 0; q < 4; ++q) {
            float4 t4 = th4[q];
            o16[q * 4 + 0] += z * t4.x;
            o16[q * 4 + 1] += z * t4.y;
            o16[q * 4 + 2] += z * t4.z;
            o16[q * 4 + 3] += z * t4.w;
        }
    }
    __syncthreads();

    // ReLU + bounce through LDS for coalesced bf16 stores
#pragma unroll
    for (int i = 0; i < 16; ++i)
        zscr[wv][(lh * 16 + i) * 16 + r] = fmaxf(o16[i], 0.f);
    __syncthreads();

    const float* rp = &zscr[wv][lane * 16];   // row o = lane, 16 columns
    v8bf s0, s1;
#pragma unroll
    for (int j = 0; j < 8; ++j) { s0[j] = (__bf16)rp[j]; s1[j] = (__bf16)rp[8 + j]; }
    __bf16* op = Out + ((size_t)bt * FOUT + lane) * NB + n0;
    *(v8bf*)op = s0;
    *((v8bf*)op + 1) = s1;
}

// ---------------------------------------------------------------------------
// Fused temporal convs (kernel (1,2), VALID, circular pad back to T, relu) x2
// Block handles (b, 32-column n tile).  xg/xt layout: [(b*T+t)*32+c][n] bf16.
// ---------------------------------------------------------------------------
__global__ __launch_bounds__(256) void k_tconv(const __bf16* __restrict__ xg,
                                               const float* __restrict__ w1,
                                               const float* __restrict__ b1,
                                               const float* __restrict__ w2,
                                               const float* __restrict__ b2,
                                               __bf16* __restrict__ xt) {
    __shared__ float s[TT_ * FOUT * 32];   // [t][c][nl]
    const int b  = blockIdx.x >> 6;
    const int n0 = (blockIdx.x & 63) * 32;
    const int tid = threadIdx.x;

    for (int i = tid; i < TT_ * FOUT * 32; i += 256) {
        const int nl = i & 31;
        const int tc = i >> 5;            // t*32 + c
        s[i] = (float)xg[(((size_t)b * TT_) * FOUT + tc) * NB + n0 + nl];
    }
    __syncthreads();

    float s1[4][TT_];
#pragma unroll
    for (int p = 0; p < 4; ++p) {
        const int id = tid + p * 256;
        const int o = id >> 5, nl = id & 31;
        float y[TT_ - 1];
#pragma unroll
        for (int t = 0; t < TT_ - 1; ++t) y[t] = 0.f;
        for (int c = 0; c < FOUT; ++c) {
            const float wa = w1[(o * FOUT + c) * 2 + 0];
            const float wb = w1[(o * FOUT + c) * 2 + 1];
#pragma unroll
            for (int t = 0; t < TT_ - 1; ++t)
                y[t] += wa * s[(t * FOUT + c) * 32 + nl] + wb * s[((t + 1) * FOUT + c) * 32 + nl];
        }
        const float bias = b1[o];
        s1[p][0] = fmaxf(y[TT_ - 2] + bias, 0.f);          // circular front-pad
#pragma unroll
        for (int t = 1; t < TT_; ++t) s1[p][t] = fmaxf(y[t - 1] + bias, 0.f);
    }
    __syncthreads();
#pragma unroll
    for (int p = 0; p < 4; ++p) {
        const int id = tid + p * 256;
        const int o = id >> 5, nl = id & 31;
#pragma unroll
        for (int t = 0; t < TT_; ++t) s[(t * FOUT + o) * 32 + nl] = s1[p][t];
    }
    __syncthreads();
#pragma unroll
    for (int p = 0; p < 4; ++p) {
        const int id = tid + p * 256;
        const int o = id >> 5, nl = id & 31;
        float y[TT_ - 1];
#pragma unroll
        for (int t = 0; t < TT_ - 1; ++t) y[t] = 0.f;
        for (int c = 0; c < FOUT; ++c) {
            const float wa = w2[(o * FOUT + c) * 2 + 0];
            const float wb = w2[(o * FOUT + c) * 2 + 1];
#pragma unroll
            for (int t = 0; t < TT_ - 1; ++t)
                y[t] += wa * s[(t * FOUT + c) * 32 + nl] + wb * s[((t + 1) * FOUT + c) * 32 + nl];
        }
        const float bias = b2[o];
#pragma unroll
        for (int t = 0; t < TT_; ++t) {
            const float v = fmaxf(((t == 0) ? y[TT_ - 2] : y[t - 1]) + bias, 0.f);
            xt[(((size_t)b * TT_ + t) * FOUT + o) * NB + n0 + nl] = (__bf16)v;
        }
    }
}

// ---------------------------------------------------------------------------
// Residual 1x1 conv + LayerNorm(n) + relu -> f32 output
// One block per (b,tt,o); reduce over n = 2048.
// ---------------------------------------------------------------------------
__global__ __launch_bounds__(256) void k_res_ln(const __bf16* __restrict__ xt,
                                                const float* __restrict__ x,
                                                const float* __restrict__ rw,
                                                const float* __restrict__ rb,
                                                const float* __restrict__ lng,
                                                const float* __restrict__ lnb,
                                                float* __restrict__ out) {
    const int id = blockIdx.x;
    const int o  = id & 31;
    const int tt = (id >> 5) % TT_;
    const int b  = id / (TT_ * FOUT);
    const size_t btt = (size_t)b * TT_ + tt;
    const int tid = threadIdx.x;

    float rwv[FIN];
#pragma unroll
    for (int f = 0; f < FIN; ++f) rwv[f] = rw[o * FIN + f];
    const float bias = rb[o];

    float vbuf[8], sum = 0.f, ssum = 0.f;
#pragma unroll
    for (int i = 0; i < 8; ++i) {
        const int n = tid + i * 256;
        float v = (float)xt[(btt * FOUT + o) * NB + n] + bias;
        const float* xp = x + (btt * FIN) * NB + n;
#pragma unroll
        for (int f = 0; f < FIN; ++f) v += xp[(size_t)f * NB] * rwv[f];
        vbuf[i] = v;
        sum += v;
        ssum += v * v;
    }
    __shared__ float r1[256], r2[256];
    r1[tid] = sum; r2[tid] = ssum;
    __syncthreads();
    for (int st = 128; st > 0; st >>= 1) {
        if (tid < st) { r1[tid] += r1[tid + st]; r2[tid] += r2[tid + st]; }
        __syncthreads();
    }
    const float mu  = r1[0] * (1.f / NB);
    const float var = r2[0] * (1.f / NB) - mu * mu;
    const float inv = rsqrtf(var + 1e-5f);
#pragma unroll
    for (int i = 0; i < 8; ++i) {
        const int n = tid + i * 256;
        const float v = (vbuf[i] - mu) * inv * lng[n] + lnb[n];
        out[(btt * FOUT + o) * NB + n] = fmaxf(v, 0.f);
    }
}

// ---------------------------------------------------------------------------
extern "C" void kernel_launch(void* const* d_in, const int* in_sizes, int n_in,
                              void* d_out, int out_size, void* d_ws, size_t ws_size,
                              hipStream_t stream) {
    const float* x    = (const float*)d_in[0];
    const float* cheb = (const float*)d_in[1];
    const float* th1  = (const float*)d_in[2];
    const float* th2  = (const float*)d_in[3];
    const float* m1w  = (const float*)d_in[4];
    const float* m1b  = (const float*)d_in[5];
    const float* m2w  = (const float*)d_in[6];
    const float* m2b  = (const float*)d_in[7];
    const float* t1w  = (const float*)d_in[8];
    const float* t1b  = (const float*)d_in[9];
    const float* t2w  = (const float*)d_in[10];
    const float* t2b  = (const float*)d_in[11];
    const float* rw   = (const float*)d_in[12];
    const float* rb   = (const float*)d_in[13];
    const float* lng  = (const float*)d_in[14];
    const float* lnb  = (const float*)d_in[15];
    float* out = (float*)d_out;

    char* ws = (char*)d_ws;
    float*  att_raw = (float*)ws;                       // 384 f32
    float*  att     = att_raw + 384;                    // 384 f32
    __bf16* chebbf  = (__bf16*)(ws + 4096);             // 3*2048*2048 bf16  (24 MB)
    __bf16* h1      = chebbf + 12582912ull;             // 32*12*16*2048 bf16 (24 MB)
    __bf16* h2      = h1 + 12582912ull;                 // 32*12*32*2048 bf16 (48 MB)
    __bf16* xg      = h2 + 25165824ull;                 // 32*12*32*2048 bf16 (48 MB)
    __bf16* xt      = xg + 25165824ull;                 // 32*12*32*2048 bf16 (48 MB)

    k_mean<<<BB * TT_, 256, 0, stream>>>(x, att_raw);
    k_semlp<<<1, 32, 0, stream>>>(att_raw, m1w, m1b, m2w, m2b, att);
    k_f32_to_bf16<<<12288, 256, 0, stream>>>(cheb, chebbf, 3145728);
    k_scale_to_bf16<<<12288, 256, 0, stream>>>(x, att, h1, 3145728);
    k_gconv<16><<<BB * TT_ * 16, 256, 0, stream>>>(h1, chebbf, th1, h2);
    k_gconv<32><<<BB * TT_ * 16, 256, 0, stream>>>(h2, chebbf, th2, xg);
    k_tconv<<<BB * 64, 256, 0, stream>>>(xg, t1w, t1b, t2w, t2b, xt);
    k_res_ln<<<BB * TT_ * FOUT, 256, 0, stream>>>(xt, x, rw, rb, lng, lnb, out);
}
